// Q_DenseNet_Daq_72275709657216
// MI455X (gfx1250) — 8-bit matrix operand order repaired and checked by reading, compiled, not run on hardware
//
#include <hip/hip_runtime.h>
#include <cstdint>


typedef __attribute__((ext_vector_type(8))) int v8i;

#define QMAXF 127.0f

__device__ __forceinline__ float amax_to_scale(unsigned u) {
  return fmaxf(__uint_as_float(u), 1e-8f) / QMAXF;
}
__device__ __forceinline__ int clamp_q(int q) {
  return q > 127 ? 127 : (q < -127 ? -127 : q);
}
__device__ __forceinline__ int imin(int a, int b) { return a < b ? a : b; }


__global__ void absmax2d_kernel(const float* __restrict__ x, long long n, int C,
                                int ld, unsigned* __restrict__ amax) {
  __shared__ float red[256];
  float m = 0.f;
  for (long long i = blockIdx.x * 256LL + threadIdx.x; i < n;
       i += (long long)gridDim.x * 256LL) {
    int c = (int)(i % C);
    long long p = i / C;
    m = fmaxf(m, fabsf(x[(size_t)p * ld + c]));
  }
  red[threadIdx.x] = m;
  __syncthreads();
  for (int s = 128; s > 0; s >>= 1) {
    if ((int)threadIdx.x < s)
      red[threadIdx.x] = fmaxf(red[threadIdx.x], red[threadIdx.x + s]);
    __syncthreads();
  }
  if (threadIdx.x == 0) atomicMax(amax, __float_as_uint(red[0]));
}

__global__ void fq_apply_kernel(float* __restrict__ x, long long n, int C, int ld,
                                const unsigned* __restrict__ amax) {
  long long i = blockIdx.x * 256LL + threadIdx.x;
  if (i >= n) return;
  float s = amax_to_scale(amax[0]);
  int c = (int)(i % C);
  long long p = i / C;
  float* px = x + (size_t)p * ld + c;
  float r = rintf(*px / s);
  r = fminf(QMAXF, fmaxf(-QMAXF, r));
  *px = r * s;
}

__global__ void quant_i8_kernel(const float* __restrict__ x, long long n, int C,
                                int Cp, int ld,
                                const unsigned* __restrict__ amax,
                                float* __restrict__ scale_out,
                                int8_t* __restrict__ q) {
  long long i = blockIdx.x * 256LL + threadIdx.x;
  if (i >= n) return;
  float s = amax_to_scale(amax[0]);
  if (i == 0) scale_out[0] = s;
  int c = (int)(i % Cp);
  long long p = i / Cp;
  int8_t v = 0;
  if (c < C) {
    float xv = x[(size_t)p * ld + c];
    v = (int8_t)clamp_q((int)rintf(xv / s));
  }
  q[i] = v;
}

__global__ void bn_fold_kernel(const float* __restrict__ gamma,
                               const float* __restrict__ beta,
                               const float* __restrict__ mean,
                               const float* __restrict__ var, int C,
                               float* __restrict__ wq, float* __restrict__ bout) {
  __shared__ float red[1024];
  int t = threadIdx.x;
  float w = 0.f, bb = 0.f;
  if (t < C) {
    w = gamma[t] / sqrtf(var[t] + 1e-5f);
    bb = beta[t] - mean[t] * w;
  }
  red[t] = fabsf(w);
  __syncthreads();
  for (int s = 512; s > 0; s >>= 1) {
    if (t < s) red[t] = fmaxf(red[t], red[t + s]);
    __syncthreads();
  }
  float sc = fmaxf(red[0], 1e-8f) / QMAXF;
  if (t < C) {
    float r = rintf(w / sc);
    r = fminf(QMAXF, fmaxf(-QMAXF, r));
    wq[t] = r * sc;
    bout[t] = bb;
  }
}

__global__ void bn_relu_absmax_kernel(const float* __restrict__ x, long long n,
                                      int C, int ld, const float* __restrict__ wq,
                                      const float* __restrict__ bb,
                                      unsigned* __restrict__ amax) {
  __shared__ float red[256];
  float m = 0.f;
  for (long long i = blockIdx.x * 256LL + threadIdx.x; i < n;
       i += (long long)gridDim.x * 256LL) {
    int c = (int)(i % C);
    long long p = i / C;
    float v = x[(size_t)p * ld + c] * wq[c] + bb[c];
    m = fmaxf(m, v);
  }
  red[threadIdx.x] = m;
  __syncthreads();
  for (int s = 128; s > 0; s >>= 1) {
    if ((int)threadIdx.x < s)
      red[threadIdx.x] = fmaxf(red[threadIdx.x], red[threadIdx.x + s]);
    __syncthreads();
  }
  if (threadIdx.x == 0) atomicMax(amax, __float_as_uint(red[0]));
}

__global__ void bn_relu_quant_kernel(const float* __restrict__ x, long long n,
                                     int C, int Cp, int ld,
                                     const float* __restrict__ wq,
                                     const float* __restrict__ bb,
                                     const unsigned* __restrict__ amax,
                                     float* __restrict__ scale_out,
                                     int8_t* __restrict__ q) {
  long long i = blockIdx.x * 256LL + threadIdx.x;
  if (i >= n) return;
  float s = amax_to_scale(amax[0]);
  if (i == 0) scale_out[0] = s;
  int c = (int)(i % Cp);
  long long p = i / Cp;
  int8_t qv = 0;
  if (c < C) {
    float v = fmaxf(x[(size_t)p * ld + c] * wq[c] + bb[c], 0.f);
    qv = (int8_t)clamp_q((int)rintf(v / s));
  }
  q[i] = qv;
}

__global__ void bn_relu_kernel(const float* __restrict__ x, long long n, int C,
                               int ld, const float* __restrict__ wq,
                               const float* __restrict__ bb,
                               float* __restrict__ out) {
  long long i = blockIdx.x * 256LL + threadIdx.x;
  if (i >= n) return;
  int c = (int)(i % C);
  long long p = i / C;
  out[i] = fmaxf(x[(size_t)p * ld + c] * wq[c] + bb[c], 0.f);
}

__global__ void quant_nchw_nhwc4_kernel(const float* __restrict__ x, int Nn,
                                        int Cin, int H, int W,
                                        const unsigned* __restrict__ amax,
                                        float* __restrict__ scale_out,
                                        int8_t* __restrict__ q) {
  long long n = (long long)Nn * H * W * 4;
  long long i = blockIdx.x * 256LL + threadIdx.x;
  if (i >= n) return;
  float s = amax_to_scale(amax[0]);
  if (i == 0) scale_out[0] = s;
  int ci = (int)(i & 3);
  long long t = i >> 2;
  int wd = (int)(t % W);
  t /= W;
  int hd = (int)(t % H);
  int img = (int)(t / H);
  int8_t v = 0;
  if (ci < Cin) {
    float xv = x[(((size_t)img * Cin + ci) * H + hd) * W + wd];
    v = (int8_t)clamp_q((int)rintf(xv / s));
  }
  q[i] = v;
}

__global__ void quant_repack_w_kernel(const float* __restrict__ w, int Co, int Ci,
                                      int kh, int kw, int Cip, int Kp,
                                      const unsigned* __restrict__ amax,
                                      float* __restrict__ scale_out,
                                      int8_t* __restrict__ q) {
  int n = Co * Kp;
  int i = blockIdx.x * 256 + threadIdx.x;
  if (i >= n) return;
  float s = amax_to_scale(amax[0]);
  if (i == 0) scale_out[0] = s;
  int kk = i % Kp;
  int o = i / Kp;
  int8_t v = 0;
  if (kk < kh * kw * Cip) {
    int ci = kk % Cip;
    int t = kk / Cip;
    int sx = t % kw;
    int r = t / kw;
    if (ci < Ci) {
      float xv = w[(((size_t)o * Ci + ci) * kh + r) * kw + sx];
      v = (int8_t)clamp_q((int)rintf(xv / s));
    }
  }
  q[i] = v;
}


__global__ void maxpool3_kernel(const float* __restrict__ in,
                                float* __restrict__ out, int Nn, int H, int W,
                                int C, int Ho, int Wo, int ldOut) {
  long long n = (long long)Nn * Ho * Wo * C;
  long long i = blockIdx.x * 256LL + threadIdx.x;
  if (i >= n) return;
  int c = (int)(i % C);
  long long pp = i / C;
  long long t = pp;
  int wo = (int)(t % Wo);
  t /= Wo;
  int ho = (int)(t % Ho);
  int img = (int)(t / Ho);
  float m = -3.4e38f;
  for (int r = 0; r < 3; r++) {
    int hi = ho * 2 - 1 + r;
    if ((unsigned)hi >= (unsigned)H) continue;
    for (int s = 0; s < 3; s++) {
      int wi = wo * 2 - 1 + s;
      if ((unsigned)wi >= (unsigned)W) continue;
      m = fmaxf(m, in[(((size_t)img * H + hi) * W + wi) * C + c]);
    }
  }
  out[(size_t)pp * ldOut + c] = m;
}

__global__ void avgpool2_kernel(const float* __restrict__ in,
                                float* __restrict__ out, int Nn, int H, int W,
                                int C, int Ho, int Wo, int ldOut) {
  long long n = (long long)Nn * Ho * Wo * C;
  long long i = blockIdx.x * 256LL + threadIdx.x;
  if (i >= n) return;
  int c = (int)(i % C);
  long long pp = i / C;
  long long t = pp;
  int wo = (int)(t % Wo);
  t /= Wo;
  int ho = (int)(t % Ho);
  int img = (int)(t / Ho);
  float s = 0.f;
  for (int r = 0; r < 2; r++)
    for (int q = 0; q < 2; q++)
      s += in[(((size_t)img * H + ho * 2 + r) * W + wo * 2 + q) * C + c];
  out[(size_t)pp * ldOut + c] = s * 0.25f;
}

__global__ void avgpool_global_kernel(const float* __restrict__ in,
                                      float* __restrict__ out, int Nn, int HW,
                                      int C) {
  int n = Nn * C;
  int i = blockIdx.x * 256 + threadIdx.x;
  if (i >= n) return;
  int c = i % C;
  int img = i / C;
  float s = 0.f;
  for (int t = 0; t < HW; t++) s += in[((size_t)img * HW + t) * C + c];
  out[i] = s / (float)HW;
}

__device__ __forceinline__ v8i load_a_frag(const int8_t* row, int off) {
  int4 d0 = *(const int4*)(row + off);
  int4 d1 = *(const int4*)(row + off + 32);
  v8i a;
  a[0] = d0.x; a[1] = d0.y; a[2] = d0.z; a[3] = d0.w;
  a[4] = d1.x; a[5] = d1.y; a[6] = d1.z; a[7] = d1.w;
  return a;
}
__device__ __forceinline__ v8i load_b_frag(const int8_t* col, int off) {
  int4 d0 = *(const int4*)(col + off);
  int4 d1 = *(const int4*)(col + off + 32);
  v8i b;
  b[0] = d0.x; b[1] = d0.y; b[2] = d0.z; b[3] = d0.w;
  b[4] = d1.x; b[5] = d1.y; b[6] = d1.z; b[7] = d1.w;
  return b;
}
#define WMMA_IU8(A, B, C) \
  __builtin_amdgcn_wmma_i32_16x16x64_iu8(true, (A), true, (B), (C), false, false)

__global__ __launch_bounds__(256) void conv_iu8_igemm_kernel(
    const int8_t* __restrict__ qw, const int8_t* __restrict__ qx,
    const int8_t* __restrict__ zpad, const float* __restrict__ sw,
    const float* __restrict__ sx, const float* __restrict__ bias,
    float* __restrict__ out, int Nn, int H, int W, int Ci, int kh, int kw,
    int stride, int pad, int Ho, int Wo, int Co, int Kp, int ldC, int co0) {
  const int lane = threadIdx.x & 31;
  const int wv = threadIdx.x >> 5;
  const int P = Nn * Ho * Wo;
  const int npair = blockIdx.x * 8 + wv;
  const int mpair = blockIdx.y;
  if (npair * 32 >= P) return;

  const int hh = lane >> 4;
  const int l16 = lane & 15;

  const int m0 = mpair * 32 + l16;
  const int m1 = m0 + 16;
  const int8_t* arow0 = qw + (size_t)imin(m0, Co - 1) * Kp;
  const int8_t* arow1 = qw + (size_t)imin(m1, Co - 1) * Kp;

  const int p0 = npair * 32 + l16;
  const int p1 = p0 + 16;
  const int pc0 = imin(p0, P - 1);
  const int pc1 = imin(p1, P - 1);

  v8i acc00 = (v8i)0, acc01 = (v8i)0, acc10 = (v8i)0, acc11 = (v8i)0;
  const int aoff = hh << 4;
  const int boff = hh << 4;

  if (kh == 1 && kw == 1) {
    const int8_t* b0 = qx + (size_t)pc0 * Kp;
    const int8_t* b1 = qx + (size_t)pc1 * Kp;
    v8i a0 = load_a_frag(arow0, aoff);
    v8i a1 = load_a_frag(arow1, aoff);
    v8i bb0 = load_b_frag(b0, boff);
    v8i bb1 = load_b_frag(b1, boff);
    for (int k0 = 64; k0 < Kp; k0 += 64) {
      v8i na0 = load_a_frag(arow0, k0 + aoff);
      v8i na1 = load_a_frag(arow1, k0 + aoff);
      v8i nb0 = load_b_frag(b0 + k0, boff);
      v8i nb1 = load_b_frag(b1 + k0, boff);
      acc00 = WMMA_IU8(a0, bb0, acc00);
      acc01 = WMMA_IU8(a0, bb1, acc01);
      acc10 = WMMA_IU8(a1, bb0, acc10);
      acc11 = WMMA_IU8(a1, bb1, acc11);
      a0 = na0; a1 = na1; bb0 = nb0; bb1 = nb1;
    }
    acc00 = WMMA_IU8(a0, bb0, acc00);
    acc01 = WMMA_IU8(a0, bb1, acc01);
    acc10 = WMMA_IU8(a1, bb0, acc10);
    acc11 = WMMA_IU8(a1, bb1, acc11);
  } else {
    const int HoWo = Ho * Wo;
    const int img0 = pc0 / HoWo, rem0 = pc0 - img0 * HoWo;
    const int img1 = pc1 / HoWo, rem1 = pc1 - img1 * HoWo;
    const int ho0 = rem0 / Wo, wo0 = rem0 - ho0 * Wo;
    const int ho1 = rem1 / Wo, wo1 = rem1 - ho1 * Wo;
    const int hb0 = ho0 * stride - pad, wb0 = wo0 * stride - pad;
    const int hb1 = ho1 * stride - pad, wb1 = wo1 * stride - pad;
    const int WCi = W * Ci;
    const int8_t* xin0 = qx + (size_t)img0 * H * WCi;
    const int8_t* xin1 = qx + (size_t)img1 * H * WCi;

    if ((Ci & 63) == 0) {
      int r = 0, s2 = 0, cib = 0;
      const int8_t* tb0;
      const int8_t* tb1;
      bool ok0, ok1;
      auto tap = [&](void) {
        int hi0 = hb0 + r, wi0 = wb0 + s2;
        int hi1 = hb1 + r, wi1 = wb1 + s2;
        ok0 = (unsigned)hi0 < (unsigned)H && (unsigned)wi0 < (unsigned)W;
        ok1 = (unsigned)hi1 < (unsigned)H && (unsigned)wi1 < (unsigned)W;
        tb0 = xin0 + (size_t)hi0 * WCi + wi0 * Ci;
        tb1 = xin1 + (size_t)hi1 * WCi + wi1 * Ci;
      };
      tap();
      v8i a0 = load_a_frag(arow0, aoff);
      v8i a1 = load_a_frag(arow1, aoff);
      v8i bb0 = load_b_frag(ok0 ? tb0 : zpad, boff);
      v8i bb1 = load_b_frag(ok1 ? tb1 : zpad, boff);
      for (int k0 = 64; k0 < Kp; k0 += 64) {
        cib += 64;
        if (cib == Ci) {
          cib = 0;
          if (++s2 == kw) { s2 = 0; ++r; }
          tap();
        }
        v8i na0 = load_a_frag(arow0, k0 + aoff);
        v8i na1 = load_a_frag(arow1, k0 + aoff);
        v8i nb0 = load_b_frag(ok0 ? tb0 + cib : zpad, boff);
        v8i nb1 = load_b_frag(ok1 ? tb1 + cib : zpad, boff);
        acc00 = WMMA_IU8(a0, bb0, acc00);
        acc01 = WMMA_IU8(a0, bb1, acc01);
        acc10 = WMMA_IU8(a1, bb0, acc10);
        acc11 = WMMA_IU8(a1, bb1, acc11);
        a0 = na0; a1 = na1; bb0 = nb0; bb1 = nb1;
      }
      acc00 = WMMA_IU8(a0, bb0, acc00);
      acc01 = WMMA_IU8(a0, bb1, acc01);
      acc10 = WMMA_IU8(a1, bb0, acc10);
      acc11 = WMMA_IU8(a1, bb1, acc11);
    } else {
      const int K = kh * kw * Ci;
      for (int k0 = 0; k0 < Kp; k0 += 64) {
        v8i a0 = load_a_frag(arow0, k0 + aoff);
        v8i a1 = load_a_frag(arow1, k0 + aoff);
        v8i bb0, bb1;
#pragma unroll
        for (int j = 0; j < 8; j++) {
          int kk = k0 + boff + ((j >> 2) << 5) + ((j & 3) << 2);
          int rs = kk / Ci;
          int ci = kk - rs * Ci;
          int r = rs / kw;
          int s2 = rs - r * kw;
          int hi0 = hb0 + r, wi0 = wb0 + s2;
          int hi1 = hb1 + r, wi1 = wb1 + s2;
          bool ok0 = (kk < K) && (unsigned)hi0 < (unsigned)H &&
                     (unsigned)wi0 < (unsigned)W;
          bool ok1 = (kk < K) && (unsigned)hi1 < (unsigned)H &&
                     (unsigned)wi1 < (unsigned)W;
          const int8_t* ap0 =
              ok0 ? xin0 + (size_t)hi0 * WCi + wi0 * Ci + ci : zpad;
          const int8_t* ap1 =
              ok1 ? xin1 + (size_t)hi1 * WCi + wi1 * Ci + ci : zpad;
          bb0[j] = *(const int*)ap0;
          bb1[j] = *(const int*)ap1;
        }
        acc00 = WMMA_IU8(a0, bb0, acc00);
        acc01 = WMMA_IU8(a0, bb1, acc01);
        acc10 = WMMA_IU8(a1, bb0, acc10);
        acc11 = WMMA_IU8(a1, bb1, acc11);
      }
    }
  }

  const float sc = sw[0] * sx[0];
#pragma unroll
  for (int v = 0; v < 8; v++) {
    int moA = mpair * 32 + v + (hh << 3);
    int moB = moA + 16;
    if (moA < Co) {
      float bA = (bias != nullptr) ? bias[moA] : 0.f;
      if (p0 < P) out[(size_t)p0 * ldC + co0 + moA] = (float)acc00[v] * sc + bA;
      if (p1 < P) out[(size_t)p1 * ldC + co0 + moA] = (float)acc01[v] * sc + bA;
    }
    if (moB < Co) {
      float bB = (bias != nullptr) ? bias[moB] : 0.f;
      if (p0 < P) out[(size_t)p0 * ldC + co0 + moB] = (float)acc10[v] * sc + bB;
      if (p1 < P) out[(size_t)p1 * ldC + co0 + moB] = (float)acc11[v] * sc + bB;
    }
  }
}


extern "C" void kernel_launch(void* const* d_in, const int* in_sizes, int n_in,
                              void* d_out, int out_size, void* d_ws,
                              size_t ws_size, hipStream_t stream) {
  (void)in_sizes;
  (void)n_in;
  (void)out_size;
  (void)ws_size;

  int cur = 0;
  auto nextf = [&]() -> const float* { return (const float*)d_in[cur++]; };

  const float* x_in = nextf();
  const float* fc_b = nextf();
  const float* fc_w = nextf();
  const float* ib_beta = nextf();
  const float* ib_gamma = nextf();
  const float* ib_mean = nextf();
  const float* ib_var = nextf();
  const float* init_w = nextf();
  const float* pb_beta = nextf();
  const float* pb_gamma = nextf();
  const float* pb_mean = nextf();
  const float* pb_var = nextf();

  size_t off = 0;
  auto wsalloc = [&](size_t bytes) -> void* {
    void* pp = (char*)d_ws + off;
    off = (off + bytes + 255) & ~(size_t)255;
    return pp;
  };
  unsigned* amaxA = (unsigned*)wsalloc(4);
  unsigned* amaxW = (unsigned*)wsalloc(4);
  float* sX = (float*)wsalloc(4);
  float* sW = (float*)wsalloc(4);
  int8_t* zpad = (int8_t*)wsalloc(2048);
  float* bnw = (float*)wsalloc(4096);
  float* bnb = (float*)wsalloc(4096);
  int8_t* qxb = (int8_t*)wsalloc(12845056);
  int8_t* qwb = (int8_t*)wsalloc(1024256);
  float* buf1 = (float*)wsalloc(51380224);
  float* buf2 = (float*)wsalloc(51380224);

  hipMemsetAsync(zpad, 0, 2048, stream);

  auto absmax = [&](const float* xx, long long P, int C, int ld, unsigned* am) {
    hipMemsetAsync(am, 0, 4, stream);
    long long n = P * (long long)C;
    int g = (int)((n + 255) / 256);
    if (g > 4096) g = 4096;
    absmax2d_kernel<<<g, 256, 0, stream>>>(xx, n, C, ld, am);
  };
  auto fqf = [&](float* xx, long long P, int C, int ld) {
    absmax(xx, P, C, ld, amaxA);
    long long n = P * (long long)C;
    fq_apply_kernel<<<(int)((n + 255) / 256), 256, 0, stream>>>(xx, n, C, ld,
                                                                amaxA);
  };
  auto bnfold = [&](const float* be, const float* ga, const float* me,
                    const float* va, int C) {
    bn_fold_kernel<<<1, 1024, 0, stream>>>(ga, be, me, va, C, bnw, bnb);
  };
  auto bnrq = [&](const float* xx, long long P, int C, int ld, int8_t* q) {
    hipMemsetAsync(amaxA, 0, 4, stream);
    long long n = P * (long long)C;
    int g = (int)((n + 255) / 256);
    if (g > 4096) g = 4096;
    bn_relu_absmax_kernel<<<g, 256, 0, stream>>>(xx, n, C, ld, bnw, bnb, amaxA);
    int Cp = (C + 63) & ~63;
    long long nq = P * (long long)Cp;
    bn_relu_quant_kernel<<<(int)((nq + 255) / 256), 256, 0, stream>>>(
        xx, nq, C, Cp, ld, bnw, bnb, amaxA, sX, q);
  };
  auto wquant = [&](const float* wp, int Co, int Ci, int kh, int kw, int Cip) {
    int nw = Co * Ci * kh * kw;
    hipMemsetAsync(amaxW, 0, 4, stream);
    int g = (nw + 255) / 256;
    if (g > 4096) g = 4096;
    absmax2d_kernel<<<g, 256, 0, stream>>>(wp, nw, nw, nw, amaxW);
    int Kp = (kh * kw * Cip + 63) & ~63;
    int nq = Co * Kp;
    quant_repack_w_kernel<<<(nq + 255) / 256, 256, 0, stream>>>(
        wp, Co, Ci, kh, kw, Cip, Kp, amaxW, sW, qwb);
  };
  auto conv = [&](const int8_t* qw_, const int8_t* qx_, const float* bias,
                  float* outp, int Nn, int H, int W, int Ci, int kh, int kw,
                  int st, int pad, int Co, int ldC, int co0) {
    int Ho = (H + 2 * pad - kh) / st + 1;
    int Wo = (W + 2 * pad - kw) / st + 1;
    int P = Nn * Ho * Wo;
    int Kp = (kh * kw * Ci + 63) & ~63;
    int ntp = (P + 31) / 32;
    dim3 grid((ntp + 7) / 8, (Co + 31) / 32);
    conv_iu8_igemm_kernel<<<grid, 256, 0, stream>>>(
        qw_, qx_, zpad, sW, sX, bias, outp, Nn, H, W, Ci, kh, kw, st, pad, Ho,
        Wo, Co, Kp, ldC, co0);
  };

  const int NB = 16;

  {
    long long nx = (long long)NB * 3 * 224 * 224;
    absmax(x_in, 1, (int)nx, (int)nx, amaxA);
    long long nq = (long long)NB * 224 * 224 * 4;
    quant_nchw_nhwc4_kernel<<<(int)((nq + 255) / 256), 256, 0, stream>>>(
        x_in, NB, 3, 224, 224, amaxA, sX, qxb);
  }

  wquant(init_w, 64, 3, 7, 7, 4);
  conv(qwb, qxb, nullptr, buf1, NB, 224, 224, 4, 7, 7, 2, 3, 64, 64, 0);
  long long P1 = (long long)NB * 112 * 112;
  fqf(buf1, P1, 64, 64);
  bnfold(ib_beta, ib_gamma, ib_mean, ib_var, 64);
  {
    long long n = P1 * 64;
    bn_relu_kernel<<<(int)((n + 255) / 256), 256, 0, stream>>>(buf1, n, 64, 64,
                                                               bnw, bnb, buf2);
  }
  long long P = (long long)NB * 56 * 56;
  {
    long long n = P * 64;
    maxpool3_kernel<<<(int)((n + 255) / 256), 256, 0, stream>>>(
        buf2, buf1, NB, 112, 112, 64, 56, 56, 256);
  }
  fqf(buf1, P, 64, 256);

  int ch = 64, Hs = 56, ld = 256;
  const int stage_ld[4] = {256, 512, 1024, 1024};
  const int nunits[4] = {6, 12, 24, 16};

  for (int s = 0; s < 4; s++) {
    if (s > 0) {
      const float* tbb = nextf();
      const float* tbg = nextf();
      const float* tbm = nextf();
      const float* tbv = nextf();
      const float* tcw = nextf();
      bnfold(tbb, tbg, tbm, tbv, ch);
      bnrq(buf1, P, ch, ld, qxb);
      int Co = ch / 2;
      wquant(tcw, Co, ch, 1, 1, ch);
      conv(qwb, qxb, nullptr, buf2, NB, Hs, Hs, ch, 1, 1, 1, 0, Co, Co, 0);
      fqf(buf2, P, Co, Co);
      int Hn = Hs / 2;
      long long Pn = (long long)NB * Hn * Hn;
      ld = stage_ld[s];
      {
        long long n = Pn * Co;
        avgpool2_kernel<<<(int)((n + 255) / 256), 256, 0, stream>>>(
            buf2, buf1, NB, Hs, Hs, Co, Hn, Hn, ld);
      }
      Hs = Hn;
      P = Pn;
      ch = Co;
      fqf(buf1, P, ch, ld);
    }
    for (int u = 0; u < nunits[s]; u++) {
      const float* b1b = nextf();
      const float* b1g = nextf();
      const float* b1m = nextf();
      const float* b1v = nextf();
      const float* b2b = nextf();
      const float* b2g = nextf();
      const float* b2m = nextf();
      const float* b2v = nextf();
      const float* c1w = nextf();
      const float* c2w = nextf();
      bnfold(b1b, b1g, b1m, b1v, ch);
      bnrq(buf1, P, ch, ld, qxb);
      wquant(c1w, 128, ch, 1, 1, ch);
      conv(qwb, qxb, nullptr, buf2, NB, Hs, Hs, ch, 1, 1, 1, 0, 128, 128, 0);
      fqf(buf2, P, 128, 128);
      bnfold(b2b, b2g, b2m, b2v, 128);
      bnrq(buf2, P, 128, 128, qxb);
      wquant(c2w, 32, 128, 3, 3, 128);
      conv(qwb, qxb, nullptr, buf1, NB, Hs, Hs, 128, 3, 3, 1, 1, 32, ld, ch);
      ch += 32;
      fqf(buf1, P, ch, ld);
    }
  }

  bnfold(pb_beta, pb_gamma, pb_mean, pb_var, 1024);
  {
    long long n = P * 1024;
    bn_relu_kernel<<<(int)((n + 255) / 256), 256, 0, stream>>>(
        buf1, n, 1024, ld, bnw, bnb, buf2);
  }
  fqf(buf2, P, 1024, 1024);
  {
    int n = NB * 1024;
    avgpool_global_kernel<<<(n + 255) / 256, 256, 0, stream>>>(buf2, buf1, NB,
                                                               49, 1024);
  }
  absmax(buf1, NB, 1024, 1024, amaxA);
  {
    int n = NB * 1024;
    quant_i8_kernel<<<(n + 255) / 256, 256, 0, stream>>>(buf1, n, 1024, 1024,
                                                         1024, amaxA, sX, qxb);
  }
  wquant(fc_w, 1000, 1024, 1, 1, 1024);
  conv(qwb, qxb, fc_b, (float*)d_out, NB, 1, 1, 1024, 1, 1, 1, 0, 1000, 1000,
       0);
}
